// EPEncoder_4947802325319
// MI455X (gfx1250) — compile-verified, matrix operand order checked by reading, not run on hardware
//
#include <hip/hip_runtime.h>
#include <math.h>

typedef __attribute__((ext_vector_type(16))) _Float16 v16h;
typedef __attribute__((ext_vector_type(8)))  _Float16 v8h;
typedef __attribute__((ext_vector_type(8)))  float    v8f;

#define PI_F  3.14159265358979323846f
#define TWO_PI_F 6.28318530717958647692f

// ---------------------------------------------------------------------------
// WMMA fragment helpers (16x16x32 f16, wave32).
// A fragment (16x32, row-major source): lane L holds row (L&15); halves 0..7 =
// K = klo..klo+7, halves 8..15 = K = 16+klo..16+klo+7 where klo = (L>>4)*8.
// B fragment loaded identically from W^T (so lane holds a column of W).
// ---------------------------------------------------------------------------
__device__ __forceinline__ v16h frag16(const _Float16* base, int stride_halfs) {
    const int lane = threadIdx.x & 31;
    const int r    = lane & 15;
    const int klo  = (lane >> 4) << 3;
    const _Float16* p = base + (size_t)r * stride_halfs + klo;
    v8h lo = *(const v8h*)p;
    v8h hi = *(const v8h*)(p + 16);
    return __builtin_shufflevector(lo, hi, 0,1,2,3,4,5,6,7,8,9,10,11,12,13,14,15);
}

__device__ __forceinline__ v8f wmma_f16(v16h a, v16h b, v8f c) {
    return __builtin_amdgcn_wmma_f32_16x16x32_f16(false, a, false, b, (short)0, c, false, false);
}

// float atomic max via integer punning (init must be -inf)
__device__ __forceinline__ void atomicMaxF(float* addr, float val) {
    if (val >= 0.0f) atomicMax((int*)addr, __float_as_int(val));
    else             atomicMin((unsigned int*)addr, __float_as_uint(val));
}

// ---------------------------------------------------------------------------
// Weight prep: W (f32, [K][Nout] row-major) -> Wt (f16, [Nout][Kpad], zero pad)
// ---------------------------------------------------------------------------
__global__ void k_prep_wt(const float* __restrict__ W, _Float16* __restrict__ Wt,
                          int K, int Nout, int Kpad) {
    int i = blockIdx.x * 256 + threadIdx.x;
    if (i >= Nout * Kpad) return;
    int n = i / Kpad, k = i % Kpad;
    Wt[i] = (k < K) ? (_Float16)W[(size_t)k * Nout + n] : (_Float16)0.0f;
}

__global__ void k_fill(float* __restrict__ p, float v, size_t n) {
    size_t i = (size_t)blockIdx.x * 256 + threadIdx.x;
    if (i < n) p[i] = v;
}

// ---------------------------------------------------------------------------
// Generic WMMA GEMM: out[M x Nout] = act(Ah[M x K](f16) @ W + bias)
// Wt is [Nout][K] f16. Block: 32 rows x 128 cols (blockIdx.y picks col chunk),
// 8 waves; wave w owns cols w*16..w*16+15 with TWO row-tiles (2 wmma per B
// fragment -> half the weight traffic per MAC). A tile staged into LDS with
// gfx1250 async-to-LDS DMA (ASYNCcnt), rows clamped instead of zero-padded
// (pad-row outputs are masked at the store). act: 0 none, 1 relu, 2 sigmoid.
// ---------------------------------------------------------------------------
__global__ __launch_bounds__(256)
void k_gemm(const _Float16* __restrict__ Ah, const _Float16* __restrict__ Wt,
            const float* __restrict__ bias, float* __restrict__ outF,
            _Float16* __restrict__ outH, int M, int K, int Nout, int act) {
    extern __shared__ __align__(16) _Float16 As[];   // 32 * (K+8)
    const int strideA = K + 8;
    const int row0 = blockIdx.x * 32;
    const int nvec = K >> 3;                         // 16B chunks per row
    // Async DMA stage: LDS[As + (r,c)] = global[Ah + clamp(row0+r)*K + c].
    // 32*nvec is always a multiple of 256 (K in {128,256,512}) -> full EXEC.
    for (int idx = threadIdx.x; idx < 32 * nvec; idx += 256) {
        int r = idx / nvec, c = (idx % nvec) << 3;
        int gr = row0 + r;
        if (gr >= M) gr = M - 1;
        unsigned lds_off = (unsigned)((r * strideA + c) * sizeof(_Float16));
        const _Float16* gaddr = Ah + (size_t)gr * K + c;
        asm volatile("global_load_async_to_lds_b128 %0, %1, off"
                     :: "v"(lds_off), "v"(gaddr) : "memory");
    }
    asm volatile("s_wait_asynccnt 0x0" ::: "memory");
    __syncthreads();

    const int wave = threadIdx.x >> 5;
    const int lane = threadIdx.x & 31;
    const int col0 = blockIdx.y * 128 + wave * 16;
    const _Float16* wbase = Wt + (size_t)col0 * K;
    v8f acc0 = {};
    v8f acc1 = {};
    v16h bcur = frag16(wbase, K);
    for (int kb = 0; kb < K; kb += 32) {
        if (kb + 32 < K)
            __builtin_prefetch(wbase + (size_t)(lane & 15) * K + kb + 64, 0, 3);
        v16h a0 = frag16(&As[kb], strideA);
        v16h a1 = frag16(&As[16 * strideA + kb], strideA);
        v16h bnext = (kb + 32 < K) ? frag16(wbase + kb + 32, K) : bcur;
        acc0 = wmma_f16(a0, bcur, acc0);
        acc1 = wmma_f16(a1, bcur, acc1);
        bcur = bnext;
    }
    const int n = col0 + (lane & 15);
    const int mb = row0 + ((lane >> 4) << 3);
    const float bi = bias ? bias[n] : 0.0f;
    for (int i = 0; i < 8; ++i) {
        int m0 = mb + i, m1 = mb + 16 + i;
        if (m0 < M) {
            float v = acc0[i] + bi;
            if (act == 1)      v = fmaxf(v, 0.0f);
            else if (act == 2) v = 1.0f / (1.0f + __expf(-v));
            if (outF) outF[(size_t)m0 * Nout + n] = v;
            if (outH) outH[(size_t)m0 * Nout + n] = (_Float16)v;
        }
        if (m1 < M) {
            float v = acc1[i] + bi;
            if (act == 1)      v = fmaxf(v, 0.0f);
            else if (act == 2) v = 1.0f / (1.0f + __expf(-v));
            if (outF) outF[(size_t)m1 * Nout + n] = v;
            if (outH) outH[(size_t)m1 * Nout + n] = (_Float16)v;
        }
    }
}

// ---------------------------------------------------------------------------
// Fourier encoder: block = 16 rows, 256 threads (8 waves).
// ---------------------------------------------------------------------------
#define FK1 160   // padded K for [cos|sin|x] = 129 features
#define FSA 168   // LDS stride (halfs) for feature tile
#define FSH 136   // LDS stride (halfs) for hidden f16 tile
#define FSB 132   // LDS stride (floats) for hidden f32 tile

__device__ __forceinline__ void tile_ln_relu(float* hbuf, _Float16* hA,
                                             const float* g, const float* b,
                                             float* red, float* mv) {
    const int t = threadIdx.x;
    const int r = t >> 4, seg = t & 15;
    float p = 0.0f;
    for (int j = 0; j < 8; ++j) p += hbuf[r * FSB + seg * 8 + j];
    red[r * 16 + seg] = p;
    __syncthreads();
    if (seg == 0) {
        float s = 0.0f;
        for (int j = 0; j < 16; ++j) s += red[r * 16 + j];
        mv[r * 2] = s * (1.0f / 128.0f);
    }
    __syncthreads();
    float mean = mv[r * 2];
    p = 0.0f;
    for (int j = 0; j < 8; ++j) { float d = hbuf[r * FSB + seg * 8 + j] - mean; p += d * d; }
    red[r * 16 + seg] = p;
    __syncthreads();
    if (seg == 0) {
        float s = 0.0f;
        for (int j = 0; j < 16; ++j) s += red[r * 16 + j];
        mv[r * 2 + 1] = rsqrtf(s * (1.0f / 128.0f) + 1e-5f);
    }
    __syncthreads();
    float rs = mv[r * 2 + 1];
    for (int j = 0; j < 8; ++j) {
        int c = seg * 8 + j;
        float y = (hbuf[r * FSB + c] - mean) * rs * g[c] + b[c];
        hA[r * FSH + c] = (_Float16)fmaxf(y, 0.0f);
    }
    __syncthreads();
}

__global__ __launch_bounds__(256)
void k_fourier(const float* __restrict__ x, int din,
               const float* __restrict__ freqs,
               const _Float16* __restrict__ w1t,   // [din][128][FK1]
               const float* __restrict__ b1, const float* __restrict__ g1,
               const float* __restrict__ b1n,
               const _Float16* __restrict__ w2t,   // [din][128][128]
               const float* __restrict__ b2,       // [din][128]
               const float* __restrict__ catsum,   // [N][128] or null
               const float* __restrict__ gout, const float* __restrict__ bout,
               const _Float16* __restrict__ woutt, // [128][128]
               const float* __restrict__ bo,
               float* __restrict__ out, int N) {
    __shared__ __align__(16) _Float16 featA[16 * FSA];
    __shared__ __align__(16) _Float16 hA[16 * FSH];
    __shared__ float hbuf[16 * FSB];
    __shared__ float red[256];
    __shared__ float mv[32];

    const int t = threadIdx.x;
    const int wave = t >> 5;
    const int lane = t & 31;
    const int row0 = blockIdx.x * 16;
    v8f acc2 = {};

    for (int d = 0; d < din; ++d) {
        for (int i = t; i < 16 * FSA; i += 256) featA[i] = (_Float16)0.0f;
        __syncthreads();
        for (int i = t; i < 16 * 64; i += 256) {
            int r = i >> 6, f = i & 63;
            int gn = row0 + r;
            float xv = (gn < N) ? x[(size_t)gn * din + d] : 0.0f;
            float xf = xv * freqs[d * 64 + f] * TWO_PI_F;
            float s, c;
            __sincosf(xf, &s, &c);
            featA[r * FSA + f]      = (_Float16)c;
            featA[r * FSA + 64 + f] = (_Float16)s;
            if (f == 0) featA[r * FSA + 128] = (_Float16)xv;
        }
        __syncthreads();
        // GEMM1: [16 x 160] x [160 -> 128]
        v8f acc1 = {};
        const _Float16* wb = w1t + (size_t)d * 128 * FK1 + (size_t)wave * 16 * FK1;
        for (int kb = 0; kb < FK1; kb += 32) {
            v16h a = frag16(&featA[kb], FSA);
            v16h b = frag16(wb + kb, FK1);
            acc1 = wmma_f16(a, b, acc1);
        }
        {
            int n  = wave * 16 + (lane & 15);
            int mb = (lane >> 4) << 3;
            float bi = b1[d * 128 + n];
            for (int i = 0; i < 8; ++i) hbuf[(mb + i) * FSB + n] = acc1[i] + bi;
        }
        __syncthreads();
        tile_ln_relu(hbuf, hA, g1 + d * 128, b1n + d * 128, red, mv);
        // GEMM2 (accumulated in registers across d)
        const _Float16* wb2 = w2t + (size_t)d * 128 * 128 + (size_t)wave * 16 * 128;
        for (int kb = 0; kb < 128; kb += 32) {
            v16h a = frag16(&hA[kb], FSH);
            v16h b = frag16(wb2 + kb, 128);
            acc2 = wmma_f16(a, b, acc2);
        }
        __syncthreads();
    }
    // epilogue: + sum_d b2 + cat_sum -> LN(gout,bout)+relu -> GEMM3 -> out
    {
        int n  = wave * 16 + (lane & 15);
        int mb = (lane >> 4) << 3;
        float bs = 0.0f;
        for (int d = 0; d < din; ++d) bs += b2[d * 128 + n];
        for (int i = 0; i < 8; ++i) {
            int r = mb + i, gn = row0 + r;
            float cs = (catsum && gn < N) ? catsum[(size_t)gn * 128 + n] : 0.0f;
            hbuf[r * FSB + n] = acc2[i] + bs + cs;
        }
    }
    __syncthreads();
    tile_ln_relu(hbuf, hA, gout, bout, red, mv);
    v8f acc3 = {};
    const _Float16* wb3 = woutt + (size_t)wave * 16 * 128;
    for (int kb = 0; kb < 128; kb += 32) {
        v16h a = frag16(&hA[kb], FSH);
        v16h b = frag16(wb3 + kb, 128);
        acc3 = wmma_f16(a, b, acc3);
    }
    {
        int n  = wave * 16 + (lane & 15);
        int mb = (lane >> 4) << 3;
        float bi = bo[n];
        for (int i = 0; i < 8; ++i) {
            int gn = row0 + mb + i;
            if (gn < N) out[(size_t)gn * 128 + n] = acc3[i] + bi;
        }
    }
}

// ---------------------------------------------------------------------------
// LayerNorm over 128 cols. One block per row, 128 threads.
// ---------------------------------------------------------------------------
__global__ __launch_bounds__(128)
void k_ln(const float* __restrict__ in, const float* __restrict__ g,
          const float* __restrict__ b, const float* __restrict__ resid,
          float* __restrict__ outF, _Float16* __restrict__ outH) {
    __shared__ float red[128];
    const int t = threadIdx.x;
    const size_t row = blockIdx.x;
    float v = in[row * 128 + t];
    red[t] = v; __syncthreads();
    for (int s = 64; s > 0; s >>= 1) { if (t < s) red[t] += red[t + s]; __syncthreads(); }
    float mean = red[0] * (1.0f / 128.0f); __syncthreads();
    float d = v - mean;
    red[t] = d * d; __syncthreads();
    for (int s = 64; s > 0; s >>= 1) { if (t < s) red[t] += red[t + s]; __syncthreads(); }
    float rs = rsqrtf(red[0] * (1.0f / 128.0f) + 1e-5f);
    float y = d * rs * g[t] + b[t];
    if (resid) y += resid[row * 128 + t];
    if (outF) outF[row * 128 + t] = y;
    if (outH) outH[row * 128 + t] = (_Float16)y;
}

// ---------------------------------------------------------------------------
// Geometry / feature kernels
// ---------------------------------------------------------------------------
__global__ void k_xp(const float* __restrict__ cps, const float* __restrict__ rpos,
                     const float* __restrict__ rhead, float* __restrict__ xp, int N) {
    int n = blockIdx.x * 256 + threadIdx.x;
    if (n >= N) return;
    const float* base = cps + (size_t)n * 24 + 16;   // [3][4][2], last group
    float s, c; __sincosf(rhead[n], &s, &c);
    float px = rpos[n * 2], py = rpos[n * 2 + 1];
    float rx[4], ry[4];
    for (int p = 0; p < 4; ++p) {
        float dx = base[p * 2] - px, dy = base[p * 2 + 1] - py;
        rx[p] =  dx * c + dy * s;
        ry[p] = -dx * s + dy * c;
    }
    for (int p = 0; p < 3; ++p) {
        xp[(size_t)n * 6 + p * 2]     = rx[p + 1] - rx[p];
        xp[(size_t)n * 6 + p * 2 + 1] = ry[p + 1] - ry[p];
    }
}

__global__ void k_xa(const float* __restrict__ cmt, float* __restrict__ xa, int N) {
    int n = blockIdx.x * 256 + threadIdx.x;
    if (n >= N) return;
    const float* b = cmt + (size_t)n * 12;
    for (int k = 0; k < 5; ++k) {
        xa[(size_t)n * 10 + k * 2]     = b[(k + 1) * 2]     - b[k * 2];
        xa[(size_t)n * 10 + k * 2 + 1] = b[(k + 1) * 2 + 1] - b[k * 2 + 1];
    }
}

__global__ void k_pose(const float* __restrict__ agent_x, float* __restrict__ pos,
                       float* __restrict__ head, int N) {
    int n = blockIdx.x * 256 + threadIdx.x;
    if (n >= N) return;
    const float* a = agent_x + (size_t)n * 150 + 147;   // [50][3], last step
    pos[n * 2] = a[0]; pos[n * 2 + 1] = a[1]; head[n] = a[2];
}

__global__ void k_relf(const float* __restrict__ ps, const float* __restrict__ os,
                       const float* __restrict__ pd, const float* __restrict__ od,
                       const int* __restrict__ src, const int* __restrict__ dst,
                       float* __restrict__ out, int E) {
    int e = blockIdx.x * 256 + threadIdx.x;
    if (e >= E) return;
    int s = src[e], d = dst[e];
    float rx = ps[s * 2] - pd[d * 2], ry = ps[s * 2 + 1] - pd[d * 2 + 1];
    float ro = os[s] - od[d];
    ro = fmodf(ro + PI_F, TWO_PI_F);
    if (ro < 0.0f) ro += TWO_PI_F;
    ro -= PI_F;
    float cs, cc; __sincosf(od[d], &cs, &cc);
    float ang = atan2f(cc * ry - cs * rx, cc * rx + cs * ry);
    out[(size_t)e * 5 + 0] = sqrtf(rx * rx + ry * ry);
    out[(size_t)e * 5 + 1] = cosf(ang);
    out[(size_t)e * 5 + 2] = sinf(ang);
    out[(size_t)e * 5 + 3] = cosf(ro);
    out[(size_t)e * 5 + 4] = sinf(ro);
}

__global__ void k_catsum_pl(const float* __restrict__ emb, const int* __restrict__ type,
                            float* __restrict__ cs, int N) {
    int i = blockIdx.x * 256 + threadIdx.x;
    if (i >= N * 128) return;
    int n = i >> 7, c = i & 127;
    cs[i] = emb[(size_t)type[n] * 128 + c];
}

__global__ __launch_bounds__(128)
void k_time_catsum(const float* __restrict__ tw, const float* __restrict__ w1,
                   const float* __restrict__ b1, const float* __restrict__ g,
                   const float* __restrict__ b, const float* __restrict__ w2,
                   const float* __restrict__ b2, const float* __restrict__ temb,
                   const int* __restrict__ otype, float* __restrict__ cs) {
    __shared__ float red[128];
    __shared__ float h[128];
    const int t = threadIdx.x;
    const int n = blockIdx.x;
    float t0 = tw[n * 2], t1 = tw[n * 2 + 1];
    float v = t0 * w1[t] + t1 * w1[128 + t] + b1[t];
    red[t] = v; __syncthreads();
    for (int s = 64; s > 0; s >>= 1) { if (t < s) red[t] += red[t + s]; __syncthreads(); }
    float mean = red[0] * (1.0f / 128.0f); __syncthreads();
    float d = v - mean;
    red[t] = d * d; __syncthreads();
    for (int s = 64; s > 0; s >>= 1) { if (t < s) red[t] += red[t + s]; __syncthreads(); }
    float rs = rsqrtf(red[0] * (1.0f / 128.0f) + 1e-5f);
    h[t] = fmaxf(d * rs * g[t] + b[t], 0.0f);
    __syncthreads();
    float o = b2[t];
    for (int j = 0; j < 128; ++j) o += h[j] * w2[j * 128 + t];
    cs[(size_t)n * 128 + t] = o + temb[(size_t)otype[n] * 128 + t];
}

// ---------------------------------------------------------------------------
// Edge attention kernels (segment softmax via atomics)
// ---------------------------------------------------------------------------
__global__ void k_edge_sim(const float* __restrict__ q, const float* __restrict__ k,
                           const float* __restrict__ kr, const int* __restrict__ src,
                           const int* __restrict__ dst, float* __restrict__ sim,
                           float* __restrict__ mbuf, int E) {
    int e = blockIdx.x * 256 + threadIdx.x;
    if (e >= E) return;
    int s = src[e], d = dst[e];
    for (int h = 0; h < 8; ++h) {
        float acc = 0.0f;
        for (int c = 0; c < 16; ++c) {
            int idx = h * 16 + c;
            acc += q[(size_t)d * 128 + idx] * (k[(size_t)s * 128 + idx] + kr[(size_t)e * 128 + idx]);
        }
        acc *= 0.25f;   // HD^-0.5
        sim[(size_t)e * 8 + h] = acc;
        atomicMaxF(&mbuf[(size_t)d * 8 + h], acc);
    }
}

__global__ void k_m_fix(float* __restrict__ m, int n) {
    int i = blockIdx.x * 256 + threadIdx.x;
    if (i >= n) return;
    float v = m[i];
    m[i] = (v > -1e38f && v < 1e38f) ? v : 0.0f;
}

__global__ void k_edge_exp(float* __restrict__ sim, const float* __restrict__ m,
                           const int* __restrict__ dst, float* __restrict__ den, int E) {
    int e = blockIdx.x * 256 + threadIdx.x;
    if (e >= E) return;
    int d = dst[e];
    for (int h = 0; h < 8; ++h) {
        float ev = __expf(sim[(size_t)e * 8 + h] - m[(size_t)d * 8 + h]);
        sim[(size_t)e * 8 + h] = ev;
        atomicAdd(&den[(size_t)d * 8 + h], ev);
    }
}

__global__ __launch_bounds__(128)
void k_edge_agg(const float* __restrict__ v, const float* __restrict__ vr,
                const float* __restrict__ ebuf, const float* __restrict__ den,
                const int* __restrict__ src, const int* __restrict__ dst,
                float* __restrict__ agg) {
    int e = blockIdx.x;
    int c = threadIdx.x;
    int s = src[e], d = dst[e];
    int h = c >> 4;
    float attn = ebuf[(size_t)e * 8 + h] / fmaxf(den[(size_t)d * 8 + h], 1e-12f);
    float vj = v[(size_t)s * 128 + c] + vr[(size_t)e * 128 + c];
    atomicAdd(&agg[(size_t)d * 128 + c], vj * attn);
}

__global__ void k_pack2(const float* __restrict__ agg, const _Float16* __restrict__ xnh,
                        _Float16* __restrict__ packed, int N) {
    int i = blockIdx.x * 256 + threadIdx.x;
    if (i >= N * 256) return;
    int n = i >> 8, c = i & 255;
    packed[i] = (c < 128) ? (_Float16)agg[(size_t)n * 128 + c]
                          : xnh[(size_t)n * 128 + (c - 128)];
}

__global__ void k_gate_o(const float* __restrict__ agg, const float* __restrict__ g,
                         const float* __restrict__ s, _Float16* __restrict__ oh, int N) {
    int i = blockIdx.x * 256 + threadIdx.x;
    if (i >= N * 128) return;
    float a = agg[i];
    oh[i] = (_Float16)(a + g[i] * (s[i] - a));
}

// ---------------------------------------------------------------------------
// Host side
// ---------------------------------------------------------------------------
struct FP { size_t b1, b1n, b2, bo, bout, freqs, g1, gout, w1, w2, wout; };

static FP fourier_ofs(size_t& o, int din) {
    FP f;
    f.b1 = o;    o += (size_t)din * 128;
    f.b1n = o;   o += (size_t)din * 128;
    f.b2 = o;    o += (size_t)din * 128;
    f.bo = o;    o += 128;
    f.bout = o;  o += 128;
    f.freqs = o; o += (size_t)din * 64;
    f.g1 = o;    o += (size_t)din * 128;
    f.gout = o;  o += 128;
    f.w1 = o;    o += (size_t)din * 129 * 128;
    f.w2 = o;    o += (size_t)din * 128 * 128;
    f.wout = o;  o += 128 * 128;
    return f;
}

struct AO { size_t b1,b2,bg,bo,bq,bs,bv,bvr,ffpost_b,ffpost_g,ffpre_b,ffpre_g,
            ln_r_b,ln_r_g,ln_x_b,ln_x_g,post_b,post_g,w1,w2,wg,wk,wkr,wo,wq,ws,wv,wvr; };

extern "C" void kernel_launch(void* const* d_in, const int* in_sizes, int n_in,
                              void* d_out, int out_size, void* d_ws, size_t ws_size,
                              hipStream_t stream) {
    (void)n_in; (void)out_size; (void)ws_size;
    const float* P        = (const float*)d_in[0];
    const float* mapel    = (const float*)d_in[1];
    const float* rpos     = (const float*)d_in[2];
    const float* rhead    = (const float*)d_in[3];
    const float* cmt      = (const float*)d_in[4];
    const float* tw       = (const float*)d_in[5];
    const float* agent_x  = (const float*)d_in[6];
    const int*   mtype    = (const int*)d_in[7];
    const int*   otype    = (const int*)d_in[8];
    const int*   ei_pp    = (const int*)d_in[9];
    const int*   ei_pa    = (const int*)d_in[10];
    const int*   ei_aa    = (const int*)d_in[11];

    const int n_pl = in_sizes[2] / 2;
    const int n_a  = in_sizes[5] / 2;
    const int E_pp = in_sizes[9] / 2;
    const int E_pa = in_sizes[10] / 2;
    const int E_aa = in_sizes[11] / 2;

    // ---- param offsets (JAX pytree leaf order: sorted dict keys) ----
    size_t po = 0;
    AO A[2];
    for (int l = 0; l < 2; ++l) {
        AO& a = A[l];
        a.b1 = po; po += 512;   a.b2 = po; po += 128;
        a.bg = po; po += 128;   a.bo = po; po += 128;
        a.bq = po; po += 128;   a.bs = po; po += 128;
        a.bv = po; po += 128;   a.bvr = po; po += 128;
        a.ffpost_b = po; po += 128; a.ffpost_g = po; po += 128;
        a.ffpre_b = po; po += 128;  a.ffpre_g = po; po += 128;
        a.ln_r_b = po; po += 128;   a.ln_r_g = po; po += 128;
        a.ln_x_b = po; po += 128;   a.ln_x_g = po; po += 128;
        a.post_b = po; po += 128;   a.post_g = po; po += 128;
        a.w1 = po; po += 128 * 512; a.w2 = po; po += 512 * 128;
        a.wg = po; po += 256 * 128; a.wk = po; po += 128 * 128;
        a.wkr = po; po += 128 * 128; a.wo = po; po += 128 * 128;
        a.wq = po; po += 128 * 128;  a.ws = po; po += 128 * 128;
        a.wv = po; po += 128 * 128;  a.wvr = po; po += 128 * 128;
    }
    FP f_a2a   = fourier_ofs(po, 5);
    FP f_pl2a  = fourier_ofs(po, 5);
    FP f_pl2pl = fourier_ofs(po, 5);
    size_t tm_b = po; po += 128; size_t tm_b1 = po; po += 128;
    size_t tm_b2 = po; po += 128; size_t tm_g = po; po += 128;
    size_t tm_w1 = po; po += 2 * 128; size_t tm_w2 = po; po += 128 * 128;
    size_t type_a = po; po += 10 * 128;
    size_t type_pl = po; po += 4 * 128;
    FP f_xa  = fourier_ofs(po, 10);
    FP f_xpl = fourier_ofs(po, 6);

    // ---- workspace bump allocator ----
    char* wsbase = (char*)d_ws;
    size_t off = 0;
    auto alloc = [&](size_t bytes) -> void* {
        void* p = wsbase + off;
        off += (bytes + 255) & ~(size_t)255;
        return p;
    };

    // ---- f16 weight prep ----
    auto prep_w = [&](size_t ofs, int K, int Nout) -> _Float16* {
        _Float16* wt = (_Float16*)alloc((size_t)Nout * K * 2);
        k_prep_wt<<<(Nout * K + 255) / 256, 256, 0, stream>>>(P + ofs, wt, K, Nout, K);
        return wt;
    };
    struct FD { _Float16 *w1t, *w2t, *woutt; };
    auto prep_fourier = [&](const FP& f, int din) -> FD {
        FD fd;
        fd.w1t   = (_Float16*)alloc((size_t)din * 128 * FK1 * 2);
        fd.w2t   = (_Float16*)alloc((size_t)din * 128 * 128 * 2);
        fd.woutt = (_Float16*)alloc(128 * 128 * 2);
        for (int d = 0; d < din; ++d) {
            k_prep_wt<<<(128 * FK1 + 255) / 256, 256, 0, stream>>>(
                P + f.w1 + (size_t)d * 129 * 128, fd.w1t + (size_t)d * 128 * FK1, 129, 128, FK1);
            k_prep_wt<<<(128 * 128 + 255) / 256, 256, 0, stream>>>(
                P + f.w2 + (size_t)d * 128 * 128, fd.w2t + (size_t)d * 128 * 128, 128, 128, 128);
        }
        k_prep_wt<<<(128 * 128 + 255) / 256, 256, 0, stream>>>(P + f.wout, fd.woutt, 128, 128, 128);
        return fd;
    };
    FD fd_xpl = prep_fourier(f_xpl, 6);
    FD fd_xa  = prep_fourier(f_xa, 10);
    FD fd_pp  = prep_fourier(f_pl2pl, 5);
    FD fd_pa  = prep_fourier(f_pl2a, 5);
    FD fd_aa  = prep_fourier(f_a2a, 5);

    _Float16 *wqt[2], *wkt[2], *wvt[2], *wst[2], *wkrt[2], *wvrt[2], *wgt[2], *wot[2], *w1t[2], *w2t[2];
    for (int l = 0; l < 2; ++l) {
        wqt[l]  = prep_w(A[l].wq, 128, 128);
        wkt[l]  = prep_w(A[l].wk, 128, 128);
        wvt[l]  = prep_w(A[l].wv, 128, 128);
        wst[l]  = prep_w(A[l].ws, 128, 128);
        wkrt[l] = prep_w(A[l].wkr, 128, 128);
        wvrt[l] = prep_w(A[l].wvr, 128, 128);
        wgt[l]  = prep_w(A[l].wg, 256, 128);
        wot[l]  = prep_w(A[l].wo, 128, 128);
        w1t[l]  = prep_w(A[l].w1, 128, 512);
        w2t[l]  = prep_w(A[l].w2, 512, 128);
    }

    // ---- activation buffers ----
    float* xp      = (float*)alloc((size_t)n_pl * 6 * 4);
    float* xa      = (float*)alloc((size_t)n_a * 10 * 4);
    float* posa    = (float*)alloc((size_t)n_a * 2 * 4);
    float* heada   = (float*)alloc((size_t)n_a * 4);
    float* relf_pp = (float*)alloc((size_t)E_pp * 5 * 4);
    float* relf_pa = (float*)alloc((size_t)E_pa * 5 * 4);
    float* relf_aa = (float*)alloc((size_t)E_aa * 5 * 4);
    float* cs_pl   = (float*)alloc((size_t)n_pl * 128 * 4);
    float* cs_a    = (float*)alloc((size_t)n_a * 128 * 4);
    float* xpl     = (float*)alloc((size_t)n_pl * 128 * 4);
    float* rpp     = (float*)alloc((size_t)E_pp * 128 * 4);
    _Float16* xnh  = (_Float16*)alloc((size_t)n_pl * 128 * 2);
    _Float16* rnh  = (_Float16*)alloc((size_t)E_pp * 128 * 2);
    float* qbuf    = (float*)alloc((size_t)n_pl * 128 * 4);
    float* kbuf    = (float*)alloc((size_t)n_pl * 128 * 4);
    float* vbuf    = (float*)alloc((size_t)n_pl * 128 * 4);
    float* sbuf    = (float*)alloc((size_t)n_pl * 128 * 4);
    float* krbuf   = (float*)alloc((size_t)E_pp * 128 * 4);
    float* vrbuf   = (float*)alloc((size_t)E_pp * 128 * 4);
    float* simb    = (float*)alloc((size_t)E_pp * 8 * 4);
    float* mbuf    = (float*)alloc((size_t)n_pl * 8 * 4);
    float* denb    = (float*)alloc((size_t)n_pl * 8 * 4);
    float* aggb    = (float*)alloc((size_t)n_pl * 128 * 4);
    _Float16* pckh = (_Float16*)alloc((size_t)n_pl * 256 * 2);
    float* gbuf    = (float*)alloc((size_t)n_pl * 128 * 4);
    _Float16* ohb  = (_Float16*)alloc((size_t)n_pl * 128 * 2);
    float* tmpb    = (float*)alloc((size_t)n_pl * 128 * 4);
    _Float16* ffh  = (_Float16*)alloc((size_t)n_pl * 128 * 2);
    _Float16* h1h  = (_Float16*)alloc((size_t)n_pl * 512 * 2);

    float* out_xpl  = (float*)d_out;
    float* out_xa   = out_xpl + (size_t)n_pl * 128;
    float* out_rpa  = out_xa + (size_t)n_a * 128;
    float* out_raa  = out_rpa + (size_t)E_pa * 128;

    auto gemm = [&](const _Float16* Ah, const _Float16* Wt, const float* bias,
                    float* outF, _Float16* outH, int M, int K, int Nout, int act) {
        dim3 g((M + 31) / 32, Nout / 128);
        size_t sh = (size_t)32 * (K + 8) * sizeof(_Float16);
        k_gemm<<<g, 256, sh, stream>>>(Ah, Wt, bias, outF, outH, M, K, Nout, act);
    };
    auto run_fourier = [&](const FP& f, const FD& fd, int din, const float* xin,
                           const float* cs, float* out, int N) {
        k_fourier<<<(N + 15) / 16, 256, 0, stream>>>(
            xin, din, P + f.freqs, fd.w1t, P + f.b1, P + f.g1, P + f.b1n,
            fd.w2t, P + f.b2, cs, P + f.gout, P + f.bout, fd.woutt, P + f.bo, out, N);
    };

    // ---- geometry / features ----
    k_xp<<<(n_pl + 255) / 256, 256, 0, stream>>>(mapel, rpos, rhead, xp, n_pl);
    k_xa<<<(n_a + 255) / 256, 256, 0, stream>>>(cmt, xa, n_a);
    k_pose<<<(n_a + 255) / 256, 256, 0, stream>>>(agent_x, posa, heada, n_a);
    k_relf<<<(E_pp + 255) / 256, 256, 0, stream>>>(rpos, rhead, rpos, rhead,
        ei_pp, ei_pp + E_pp, relf_pp, E_pp);
    k_relf<<<(E_pa + 255) / 256, 256, 0, stream>>>(rpos, rhead, posa, heada,
        ei_pa, ei_pa + E_pa, relf_pa, E_pa);
    k_relf<<<(E_aa + 255) / 256, 256, 0, stream>>>(posa, heada, posa, heada,
        ei_aa, ei_aa + E_aa, relf_aa, E_aa);
    k_catsum_pl<<<((size_t)n_pl * 128 + 255) / 256, 256, 0, stream>>>(P + type_pl, mtype, cs_pl, n_pl);
    k_time_catsum<<<n_a, 128, 0, stream>>>(tw, P + tm_w1, P + tm_b1, P + tm_g, P + tm_b,
        P + tm_w2, P + tm_b2, P + type_a, otype, cs_a);

    // ---- Fourier encoders (WMMA) ----
    run_fourier(f_xpl,   fd_xpl, 6,  xp,      cs_pl, xpl,     n_pl);
    run_fourier(f_xa,    fd_xa, 10,  xa,      cs_a,  out_xa,  n_a);
    run_fourier(f_pl2pl, fd_pp,  5,  relf_pp, nullptr, rpp,   E_pp);
    run_fourier(f_pl2a,  fd_pa,  5,  relf_pa, nullptr, out_rpa, E_pa);
    run_fourier(f_a2a,   fd_aa,  5,  relf_aa, nullptr, out_raa, E_aa);

    // ---- attention layers on x_pl / r_pl2pl ----
    for (int l = 0; l < 2; ++l) {
        const AO& a = A[l];
        k_ln<<<n_pl, 128, 0, stream>>>(xpl, P + a.ln_x_g, P + a.ln_x_b, nullptr, nullptr, xnh);
        k_ln<<<E_pp, 128, 0, stream>>>(rpp, P + a.ln_r_g, P + a.ln_r_b, nullptr, nullptr, rnh);
        gemm(xnh, wqt[l], P + a.bq, qbuf, nullptr, n_pl, 128, 128, 0);
        gemm(xnh, wkt[l], nullptr,  kbuf, nullptr, n_pl, 128, 128, 0);
        gemm(xnh, wvt[l], P + a.bv, vbuf, nullptr, n_pl, 128, 128, 0);
        gemm(xnh, wst[l], P + a.bs, sbuf, nullptr, n_pl, 128, 128, 0);
        gemm(rnh, wkrt[l], nullptr,   krbuf, nullptr, E_pp, 128, 128, 0);
        gemm(rnh, wvrt[l], P + a.bvr, vrbuf, nullptr, E_pp, 128, 128, 0);
        k_fill<<<((size_t)n_pl * 8 + 255) / 256, 256, 0, stream>>>(mbuf, -__builtin_inff(), (size_t)n_pl * 8);
        k_fill<<<((size_t)n_pl * 8 + 255) / 256, 256, 0, stream>>>(denb, 0.0f, (size_t)n_pl * 8);
        k_fill<<<((size_t)n_pl * 128 + 255) / 256, 256, 0, stream>>>(aggb, 0.0f, (size_t)n_pl * 128);
        k_edge_sim<<<(E_pp + 255) / 256, 256, 0, stream>>>(qbuf, kbuf, krbuf, ei_pp, ei_pp + E_pp, simb, mbuf, E_pp);
        k_m_fix<<<(n_pl * 8 + 255) / 256, 256, 0, stream>>>(mbuf, n_pl * 8);
        k_edge_exp<<<(E_pp + 255) / 256, 256, 0, stream>>>(simb, mbuf, ei_pp + E_pp, denb, E_pp);
        k_edge_agg<<<E_pp, 128, 0, stream>>>(vbuf, vrbuf, simb, denb, ei_pp, ei_pp + E_pp, aggb);
        k_pack2<<<((size_t)n_pl * 256 + 255) / 256, 256, 0, stream>>>(aggb, xnh, pckh, n_pl);
        gemm(pckh, wgt[l], P + a.bg, gbuf, nullptr, n_pl, 256, 128, 2);
        k_gate_o<<<((size_t)n_pl * 128 + 255) / 256, 256, 0, stream>>>(aggb, gbuf, sbuf, ohb, n_pl);
        gemm(ohb, wot[l], P + a.bo, tmpb, nullptr, n_pl, 128, 128, 0);
        k_ln<<<n_pl, 128, 0, stream>>>(tmpb, P + a.post_g, P + a.post_b, xpl, xpl, nullptr);
        k_ln<<<n_pl, 128, 0, stream>>>(xpl, P + a.ffpre_g, P + a.ffpre_b, nullptr, nullptr, ffh);
        gemm(ffh, w1t[l], P + a.b1, nullptr, h1h, n_pl, 128, 512, 1);
        gemm(h1h, w2t[l], P + a.b2, tmpb, nullptr, n_pl, 512, 128, 0);
        k_ln<<<n_pl, 128, 0, stream>>>(tmpb, P + a.ffpost_g, P + a.ffpost_b, xpl, xpl, nullptr);
    }

    hipMemcpyAsync(out_xpl, xpl, (size_t)n_pl * 128 * 4, hipMemcpyDeviceToDevice, stream);
}
